// PionStopRegressor_14645838479737
// MI455X (gfx1250) — matrix operand order repaired and checked by reading, not compiled, not run on hardware
//
#include <hip/hip_runtime.h>
#include <hip/hip_bf16.h>
#include <math.h>

typedef __attribute__((ext_vector_type(16))) _Float16 v16h;
typedef __attribute__((ext_vector_type(8)))  float    v8f;

#define NN 100000
#define EE 400000
#define GG 1024
#define HH 128
#define HD_ 4
#define DH_ 32
#define LL 3
#define FF_ 256
#define JK_ 384
#define SCALE_ 0.17677669529663687f   // 1/sqrt(32)

// ---------------------------------------------------------------------------
// float atomic max via monotonic int mapping (init memory to -inf)
// ---------------------------------------------------------------------------
__device__ inline void atomicMaxFloat(float* addr, float value) {
  if (value >= 0.0f)
    atomicMax((int*)addr, __float_as_int(value));
  else
    atomicMin((unsigned int*)addr, __float_as_uint(value));
}

// ---------------------------------------------------------------------------
// fill
// ---------------------------------------------------------------------------
__global__ void fill_kernel(float* __restrict__ p, float v, size_t n) {
  size_t i = (size_t)blockIdx.x * blockDim.x + threadIdx.x;
  if (i < n) p[i] = v;
}

// ---------------------------------------------------------------------------
// Pack f32 weight W[K][Ncols] (row major) into WMMA 16x16x32 f16 B-fragment
// layout: halves[(((nt*Kt + kt)*32 + lane)*16 + h)]
//   lane l -> col = nt*16 + (l&15);  K = kt*32 + (l>>4)*16 + h
// ---------------------------------------------------------------------------
__global__ void pack_weight_kernel(const float* __restrict__ W,
                                   _Float16* __restrict__ out,
                                   int K, int Ncols) {
  int Kt = K >> 5;
  int Nt = Ncols >> 4;
  int total = Nt * Kt * 32;
  int idx = blockIdx.x * blockDim.x + threadIdx.x;
  if (idx >= total) return;
  int lane = idx & 31;
  int kt = (idx >> 5) % Kt;
  int nt = (idx >> 5) / Kt;
  int col = nt * 16 + (lane & 15);
  int kb  = kt * 32 + ((lane >> 4) << 4);
  _Float16* dst = out + (size_t)idx * 16;
#pragma unroll
  for (int h = 0; h < 16; ++h)
    dst[h] = (_Float16)W[(size_t)(kb + h) * Ncols + col];
}

// ---------------------------------------------------------------------------
// Multi-weight WMMA GEMM:  C[m][rows, Ncols] = act(A[rows, K] @ B[m] + bias[m])
// for m in 0..NMAT-1 (NMAT=4 fuses Q/K/V/skip: A staged & converted once).
// Block = 256 threads (8 wave32), tile = 32 rows x Ncols (two 16-row M-tiles
// so each B fragment load feeds TWO v_wmma issues -> halves L2 B-traffic).
// A tile f32->f16 staged in LDS in A-fragment layout, contiguous k order (same as packed B):
//   lane l (row = l&15, grp = l>>4):
//     halves 0..7  -> K = kt*32 + h          + grp*16
//     halves 8..15 -> K = kt*32 + h          + grp*16
// ---------------------------------------------------------------------------
struct GemmArgs {
  const _Float16* B[4];
  const float*    bias[4];
  float*          C[4];
};

template <int NMAT>
__global__ __launch_bounds__(256)
void gemm_multi_kernel(const float* __restrict__ A, GemmArgs args,
                       int K, int Ncols, int act) {
  __shared__ _Float16 ldsA[JK_ * 32];   // 2 M-tiles, K up to 384 (24 KB)
  const int Kt = K >> 5;
  const int Nt = Ncols >> 4;
  const int row0 = blockIdx.x * 32;
  const int tid = threadIdx.x;
  const int lane = tid & 31;
  const int wave = tid >> 5;

  // stage A fragments for both 16-row tiles into LDS (f32 -> f16 convert)
  for (int slot = tid; slot < Kt * 64; slot += 256) {
    int kt = slot >> 6;
    int ts = (slot >> 5) & 1;
    int l  = slot & 31;
    int grp = l >> 4;
    const float* arow =
        A + (size_t)(row0 + ts * 16 + (l & 15)) * K + kt * 32 + grp * 16;
    _Float16* dst = &ldsA[slot * 16];
#pragma unroll
    for (int h = 0; h < 8; ++h) dst[h]     = (_Float16)arow[h];
#pragma unroll
    for (int h = 0; h < 8; ++h) dst[8 + h] = (_Float16)arow[8 + h];
  }
  __syncthreads();

  for (int nt = wave; nt < Nt; nt += 8) {
    if (nt + 8 < Nt)  // prefetch next column-tile's fragments into L2
      __builtin_prefetch(
          (const v16h*)args.B[0] + ((size_t)(nt + 8) * Kt) * 32 + lane, 0, 0);
    v8f acc[NMAT][2] = {};
    for (int kt = 0; kt < Kt; ++kt) {
      v16h a0 = *(const v16h*)&ldsA[((kt * 2 + 0) * 32 + lane) * 16];
      v16h a1 = *(const v16h*)&ldsA[((kt * 2 + 1) * 32 + lane) * 16];
#pragma unroll
      for (int m = 0; m < NMAT; ++m) {
        v16h b = ((const v16h*)args.B[m])[((size_t)nt * Kt + kt) * 32 + lane];
        acc[m][0] = __builtin_amdgcn_wmma_f32_16x16x32_f16(
            false, a0, false, b, (short)0, acc[m][0], false, false);
        acc[m][1] = __builtin_amdgcn_wmma_f32_16x16x32_f16(
            false, a1, false, b, (short)0, acc[m][1], false, false);
      }
    }
    int col = nt * 16 + (lane & 15);
#pragma unroll
    for (int m = 0; m < NMAT; ++m) {
      float bv = args.bias[m] ? args.bias[m][col] : 0.0f;
#pragma unroll
      for (int ts = 0; ts < 2; ++ts) {
        int rbase = row0 + ts * 16 + ((lane >> 4) << 3);
#pragma unroll
        for (int r = 0; r < 8; ++r) {
          float v = acc[m][ts][r] + bv;
          if (act == 1) v = fmaxf(v, 0.0f);
          args.C[m][(size_t)(rbase + r) * Ncols + col] = v;
        }
      }
    }
  }
}

// ---------------------------------------------------------------------------
// Input projection: h = x @ Win + b_in   (IN=4 -> bandwidth bound)
// ---------------------------------------------------------------------------
__global__ __launch_bounds__(256)
void input_proj_kernel(const float* __restrict__ x,
                       const float* __restrict__ Win,
                       const float* __restrict__ b_in,
                       float* __restrict__ h) {
  int n = blockIdx.x * 8 + (threadIdx.x >> 5);
  if (n >= NN) return;
  int lane = threadIdx.x & 31;
  float x0 = x[(size_t)n * 4 + 0], x1 = x[(size_t)n * 4 + 1];
  float x2 = x[(size_t)n * 4 + 2], x3 = x[(size_t)n * 4 + 3];
#pragma unroll
  for (int j = 0; j < 4; ++j) {
    int d = lane * 4 + j;
    float v = x0 * Win[d] + x1 * Win[HH + d] + x2 * Win[2 * HH + d]
            + x3 * Win[3 * HH + d] + b_in[d];
    h[(size_t)n * HH + d] = v;
  }
}

// ---------------------------------------------------------------------------
// Edge pass 1: logits[e,hd] = dot(q[dst,hd], k[src,hd]+e[hd]) * SCALE
// atomic segment-max into m[dst,hd].   wave per edge, 8 lanes per head.
// ---------------------------------------------------------------------------
__global__ __launch_bounds__(256)
void edge_logits_kernel(const int* __restrict__ ei,
                        const float* __restrict__ eattr,
                        const float* __restrict__ Wel,   // [4,128]
                        const float* __restrict__ q,
                        const float* __restrict__ kn,
                        float* __restrict__ logits,
                        float* __restrict__ m) {
  int e = blockIdx.x * 8 + (threadIdx.x >> 5);
  if (e >= EE) return;
  int lane = threadIdx.x & 31;
  int src = ei[e];
  int dst = ei[EE + e];
  float a0 = eattr[(size_t)e * 4 + 0], a1 = eattr[(size_t)e * 4 + 1];
  float a2 = eattr[(size_t)e * 4 + 2], a3 = eattr[(size_t)e * 4 + 3];
  float part = 0.0f;
  int d0 = lane * 4;
#pragma unroll
  for (int j = 0; j < 4; ++j) {
    int d = d0 + j;
    float ed = a0 * Wel[d] + a1 * Wel[HH + d] + a2 * Wel[2 * HH + d]
             + a3 * Wel[3 * HH + d];
    float ke = kn[(size_t)src * HH + d] + ed;
    part += q[(size_t)dst * HH + d] * ke;
  }
  part += __shfl_xor(part, 1, 32);
  part += __shfl_xor(part, 2, 32);
  part += __shfl_xor(part, 4, 32);
  if ((lane & 7) == 0) {
    int hd = lane >> 3;
    float lg = part * SCALE_;
    logits[(size_t)e * HD_ + hd] = lg;
    atomicMaxFloat(&m[(size_t)dst * HD_ + hd], lg);
  }
}

// ---------------------------------------------------------------------------
// Edge pass 2: p = exp(logit - m[dst]); s[dst] += p; agg[dst] += p * ve
// (alpha normalization by s is deferred to the node kernel)
// ---------------------------------------------------------------------------
__global__ __launch_bounds__(256)
void edge_accum_kernel(const int* __restrict__ ei,
                       const float* __restrict__ eattr,
                       const float* __restrict__ Wel,
                       const float* __restrict__ vn,
                       const float* __restrict__ logits,
                       const float* __restrict__ m,
                       float* __restrict__ s,
                       float* __restrict__ agg) {
  int e = blockIdx.x * 8 + (threadIdx.x >> 5);
  if (e >= EE) return;
  int lane = threadIdx.x & 31;
  int src = ei[e];
  int dst = ei[EE + e];
  int hd = lane >> 3;
  float p = __expf(logits[(size_t)e * HD_ + hd] - m[(size_t)dst * HD_ + hd]);
  if ((lane & 7) == 0) atomicAdd(&s[(size_t)dst * HD_ + hd], p);
  float a0 = eattr[(size_t)e * 4 + 0], a1 = eattr[(size_t)e * 4 + 1];
  float a2 = eattr[(size_t)e * 4 + 2], a3 = eattr[(size_t)e * 4 + 3];
  int d0 = lane * 4;
#pragma unroll
  for (int j = 0; j < 4; ++j) {
    int d = d0 + j;
    float ed = a0 * Wel[d] + a1 * Wel[HH + d] + a2 * Wel[2 * HH + d]
             + a3 * Wel[3 * HH + d];
    float ve = vn[(size_t)src * HH + d] + ed;
    atomicAdd(&agg[(size_t)dst * HH + d], p * ve);
  }
}

// ---------------------------------------------------------------------------
// attn = agg/(s+eps) + skip;  h = LN(h + attn)   (wave32 per node)
// ---------------------------------------------------------------------------
__global__ __launch_bounds__(256)
void attn_ln1_kernel(const float* __restrict__ hin,
                     const float* __restrict__ agg,
                     const float* __restrict__ s,
                     const float* __restrict__ skip,
                     const float* __restrict__ g,
                     const float* __restrict__ b,
                     float* __restrict__ hout) {
  int n = blockIdx.x * 8 + (threadIdx.x >> 5);
  if (n >= NN) return;
  int lane = threadIdx.x & 31;
  float t[4];
  float sum = 0.0f;
#pragma unroll
  for (int j = 0; j < 4; ++j) {
    int d = lane * 4 + j;
    int hd = d >> 5;
    float attn = agg[(size_t)n * HH + d] / (s[(size_t)n * HD_ + hd] + 1e-16f)
               + skip[(size_t)n * HH + d];
    t[j] = hin[(size_t)n * HH + d] + attn;
    sum += t[j];
  }
  for (int o = 1; o < 32; o <<= 1) sum += __shfl_xor(sum, o, 32);
  float mean = sum * (1.0f / 128.0f);
  float var = 0.0f;
#pragma unroll
  for (int j = 0; j < 4; ++j) { float dd = t[j] - mean; var += dd * dd; }
  for (int o = 1; o < 32; o <<= 1) var += __shfl_xor(var, o, 32);
  var *= (1.0f / 128.0f);
  float inv = rsqrtf(var + 1e-5f);
#pragma unroll
  for (int j = 0; j < 4; ++j) {
    int d = lane * 4 + j;
    hout[(size_t)n * HH + d] = (t[j] - mean) * inv * g[d] + b[d];
  }
}

// ---------------------------------------------------------------------------
// h = LN(h + ff2);  also writes JK slice xc[:, layer*128 : +128]
// ---------------------------------------------------------------------------
__global__ __launch_bounds__(256)
void ln2_jk_kernel(const float* __restrict__ hin,
                   const float* __restrict__ ff2,
                   const float* __restrict__ g,
                   const float* __restrict__ b,
                   float* __restrict__ hout,
                   float* __restrict__ xc, int layer) {
  int n = blockIdx.x * 8 + (threadIdx.x >> 5);
  if (n >= NN) return;
  int lane = threadIdx.x & 31;
  float t[4];
  float sum = 0.0f;
#pragma unroll
  for (int j = 0; j < 4; ++j) {
    int d = lane * 4 + j;
    t[j] = hin[(size_t)n * HH + d] + ff2[(size_t)n * HH + d];
    sum += t[j];
  }
  for (int o = 1; o < 32; o <<= 1) sum += __shfl_xor(sum, o, 32);
  float mean = sum * (1.0f / 128.0f);
  float var = 0.0f;
#pragma unroll
  for (int j = 0; j < 4; ++j) { float dd = t[j] - mean; var += dd * dd; }
  for (int o = 1; o < 32; o <<= 1) var += __shfl_xor(var, o, 32);
  var *= (1.0f / 128.0f);
  float inv = rsqrtf(var + 1e-5f);
#pragma unroll
  for (int j = 0; j < 4; ++j) {
    int d = lane * 4 + j;
    float v = (t[j] - mean) * inv * g[d] + b[d];
    hout[(size_t)n * HH + d] = v;
    xc[(size_t)n * JK_ + layer * HH + d] = v;
  }
}

// ---------------------------------------------------------------------------
// gate = ghid @ Wg2 + bg2; atomic segment max into gm[batch]
// ---------------------------------------------------------------------------
__global__ __launch_bounds__(256)
void gate_kernel(const float* __restrict__ ghid,
                 const float* __restrict__ Wg2,
                 const float* __restrict__ bg2,
                 const int* __restrict__ batch,
                 float* __restrict__ gate,
                 float* __restrict__ gm) {
  int n = blockIdx.x * 8 + (threadIdx.x >> 5);
  if (n >= NN) return;
  int lane = threadIdx.x & 31;
  float acc = 0.0f;
#pragma unroll
  for (int j = 0; j < 4; ++j) {
    int d = lane * 4 + j;
    acc += ghid[(size_t)n * HH + d] * Wg2[d];
  }
  for (int o = 1; o < 32; o <<= 1) acc += __shfl_xor(acc, o, 32);
  if (lane == 0) {
    float gv = acc + bg2[0];
    gate[n] = gv;
    atomicMaxFloat(&gm[batch[n]], gv);
  }
}

// ---------------------------------------------------------------------------
// pooling: gp = exp(gate - gm[b]); gs[b]+=gp; pooled[b,:] += gp*xc[n,:]
// ---------------------------------------------------------------------------
__global__ __launch_bounds__(256)
void pool_kernel(const float* __restrict__ gate,
                 const int* __restrict__ batch,
                 const float* __restrict__ gm,
                 float* __restrict__ gs,
                 const float* __restrict__ xc,
                 float* __restrict__ pooled) {
  int n = blockIdx.x * 8 + (threadIdx.x >> 5);
  if (n >= NN) return;
  int lane = threadIdx.x & 31;
  int b = batch[n];
  float gp = __expf(gate[n] - gm[b]);
  if (lane == 0) atomicAdd(&gs[b], gp);
  for (int d = lane; d < JK_; d += 32)
    atomicAdd(&pooled[(size_t)b * JK_ + d], gp * xc[(size_t)n * JK_ + d]);
}

// ---------------------------------------------------------------------------
// Final head: out = relu([pooled/gs, u] @ Wh1 + bh1) @ Wh2 + bh2
// one 128-thread block per graph
// ---------------------------------------------------------------------------
__global__ __launch_bounds__(128)
void head_kernel(const float* __restrict__ pooled,
                 const float* __restrict__ gs,
                 const float* __restrict__ u,
                 const float* __restrict__ Wh1, const float* __restrict__ bh1,
                 const float* __restrict__ Wh2, const float* __restrict__ bh2,
                 float* __restrict__ out) {
  int g = blockIdx.x;
  __shared__ float in_s[JK_ + 1];
  __shared__ float hid[HH];
  int t = threadIdx.x;
  float inv = 1.0f / (gs[g] + 1e-16f);
  for (int k = t; k < JK_; k += 128) in_s[k] = pooled[(size_t)g * JK_ + k] * inv;
  if (t == 0) in_s[JK_] = u[g];
  __syncthreads();
  float acc = bh1[t];
  for (int k = 0; k < JK_ + 1; ++k) acc += in_s[k] * Wh1[(size_t)k * HH + t];
  hid[t] = fmaxf(acc, 0.0f);
  __syncthreads();
  if (t < 3) {
    float o = bh2[t];
    for (int j = 0; j < HH; ++j) o += hid[j] * Wh2[j * 3 + t];
    out[(size_t)g * 3 + t] = o;
  }
}

// ---------------------------------------------------------------------------
// host
// ---------------------------------------------------------------------------
static inline size_t align256(size_t x) { return (x + 255) & ~(size_t)255; }

extern "C" void kernel_launch(void* const* d_in, const int* in_sizes, int n_in,
                              void* d_out, int out_size, void* d_ws, size_t ws_size,
                              hipStream_t stream) {
  const float* x        = (const float*)d_in[0];
  const int*   eidx     = (const int*)d_in[1];
  const float* eattr    = (const float*)d_in[2];
  const int*   batch    = (const int*)d_in[3];
  const float* u        = (const float*)d_in[4];
  const float* Win      = (const float*)d_in[5];
  const float* b_in     = (const float*)d_in[6];
  const float* Wq       = (const float*)d_in[7];
  const float* bq       = (const float*)d_in[8];
  const float* Wk       = (const float*)d_in[9];
  const float* bk       = (const float*)d_in[10];
  const float* Wv       = (const float*)d_in[11];
  const float* bv       = (const float*)d_in[12];
  const float* We       = (const float*)d_in[13];
  const float* Wskip    = (const float*)d_in[14];
  const float* bskip    = (const float*)d_in[15];
  const float* ln1_g    = (const float*)d_in[16];
  const float* ln1_b    = (const float*)d_in[17];
  const float* Wf1      = (const float*)d_in[18];
  const float* bf1      = (const float*)d_in[19];
  const float* Wf2      = (const float*)d_in[20];
  const float* bf2      = (const float*)d_in[21];
  const float* ln2_g    = (const float*)d_in[22];
  const float* ln2_b    = (const float*)d_in[23];
  const float* Wg1      = (const float*)d_in[24];
  const float* bg1      = (const float*)d_in[25];
  const float* Wg2      = (const float*)d_in[26];
  const float* bg2      = (const float*)d_in[27];
  const float* Wh1      = (const float*)d_in[28];
  const float* bh1      = (const float*)d_in[29];
  const float* Wh2      = (const float*)d_in[30];
  const float* bh2      = (const float*)d_in[31];
  float* out = (float*)d_out;

  // ---- workspace layout ----
  char* base = (char*)d_ws;
  size_t off = 0;
  auto grab = [&](size_t bytes) { char* p = base + off; off += align256(bytes); return p; };

  float* h      = (float*)grab((size_t)NN * HH * 4);
  float* qb     = (float*)grab((size_t)NN * HH * 4);
  float* knb    = (float*)grab((size_t)NN * HH * 4);
  float* vnb    = (float*)grab((size_t)NN * HH * 4);
  float* skipb  = (float*)grab((size_t)NN * HH * 4);
  float* aggb   = (float*)grab((size_t)NN * HH * 4);
  float* ffb    = (float*)grab((size_t)NN * FF_ * 4);
  float* xcb    = (float*)grab((size_t)NN * JK_ * 4);
  float* logitb = (float*)grab((size_t)EE * HD_ * 4);
  float* mb     = (float*)grab((size_t)NN * HD_ * 4);
  float* sb     = (float*)grab((size_t)NN * HD_ * 4);
  float* gateb  = (float*)grab((size_t)NN * 4);
  float* gmb    = (float*)grab((size_t)GG * 4);
  float* gsb    = (float*)grab((size_t)GG * 4);
  float* poolb  = (float*)grab((size_t)GG * JK_ * 4);
  _Float16* WqP[LL], *WkP[LL], *WvP[LL], *WsP[LL], *Wf1P[LL], *Wf2P[LL];
  for (int l = 0; l < LL; ++l) {
    WqP[l]  = (_Float16*)grab((size_t)HH * HH * 2);
    WkP[l]  = (_Float16*)grab((size_t)HH * HH * 2);
    WvP[l]  = (_Float16*)grab((size_t)HH * HH * 2);
    WsP[l]  = (_Float16*)grab((size_t)HH * HH * 2);
    Wf1P[l] = (_Float16*)grab((size_t)HH * FF_ * 2);
    Wf2P[l] = (_Float16*)grab((size_t)FF_ * HH * 2);
  }
  _Float16* Wg1P = (_Float16*)grab((size_t)JK_ * HH * 2);
  float* ff2b = qb;    // reuse: q dead after edge pass 1
  float* ghid = knb;   // reuse: k dead after edge pass 2 (gate is post-layers)

  const int NB32  = NN / 32;            // gemm row-tiles (100000 % 32 == 0)
  const int NWAVE = NN / 8;             // node wave-kernels
  const int EWAVE = EE / 8;             // edge wave-kernels

  // ---- pack weights to f16 fragment layout ----
  auto pack = [&](const float* W, _Float16* P, int K, int Nc) {
    int total = (K >> 5) * (Nc >> 4) * 32;
    hipLaunchKernelGGL(pack_weight_kernel, dim3((total + 255) / 256), dim3(256),
                       0, stream, W, P, K, Nc);
  };
  for (int l = 0; l < LL; ++l) {
    pack(Wq + (size_t)l * HH * HH, WqP[l], HH, HH);
    pack(Wk + (size_t)l * HH * HH, WkP[l], HH, HH);
    pack(Wv + (size_t)l * HH * HH, WvP[l], HH, HH);
    pack(Wskip + (size_t)l * HH * HH, WsP[l], HH, HH);
    pack(Wf1 + (size_t)l * HH * FF_, Wf1P[l], HH, FF_);
    pack(Wf2 + (size_t)l * FF_ * HH, Wf2P[l], FF_, HH);
  }
  pack(Wg1, Wg1P, JK_, HH);

  // ---- input projection ----
  hipLaunchKernelGGL(input_proj_kernel, dim3(NWAVE), dim3(256), 0, stream,
                     x, Win, b_in, h);

  auto gemm1 = [&](const float* A, const _Float16* Bp, const float* bias,
                   float* C, int K, int Nc, int act) {
    GemmArgs ga{};
    ga.B[0] = Bp; ga.bias[0] = bias; ga.C[0] = C;
    hipLaunchKernelGGL(gemm_multi_kernel<1>, dim3(NB32), dim3(256), 0, stream,
                       A, ga, K, Nc, act);
  };
  auto fill = [&](float* p, float v, size_t n) {
    hipLaunchKernelGGL(fill_kernel, dim3((unsigned)((n + 255) / 256)), dim3(256),
                       0, stream, p, v, n);
  };

  // ---- transformer layers ----
  for (int l = 0; l < LL; ++l) {
    const float* Wel = We + (size_t)l * 4 * HH;

    // fused Q/K/V/skip projection: A staged once, 4 WMMA weight sets
    GemmArgs ga{};
    ga.B[0] = WqP[l]; ga.bias[0] = bq + l * HH;    ga.C[0] = qb;
    ga.B[1] = WkP[l]; ga.bias[1] = bk + l * HH;    ga.C[1] = knb;
    ga.B[2] = WvP[l]; ga.bias[2] = bv + l * HH;    ga.C[2] = vnb;
    ga.B[3] = WsP[l]; ga.bias[3] = bskip + l * HH; ga.C[3] = skipb;
    hipLaunchKernelGGL(gemm_multi_kernel<4>, dim3(NB32), dim3(256), 0, stream,
                       h, ga, HH, HH, 0);

    fill(mb, -INFINITY, (size_t)NN * HD_);
    fill(sb, 0.0f, (size_t)NN * HD_);
    fill(aggb, 0.0f, (size_t)NN * HH);

    hipLaunchKernelGGL(edge_logits_kernel, dim3(EWAVE), dim3(256), 0, stream,
                       eidx, eattr, Wel, qb, knb, logitb, mb);
    hipLaunchKernelGGL(edge_accum_kernel, dim3(EWAVE), dim3(256), 0, stream,
                       eidx, eattr, Wel, vnb, logitb, mb, sb, aggb);

    hipLaunchKernelGGL(attn_ln1_kernel, dim3(NWAVE), dim3(256), 0, stream,
                       h, aggb, sb, skipb, ln1_g + l * HH, ln1_b + l * HH, h);

    gemm1(h, Wf1P[l], bf1 + l * FF_, ffb, HH, FF_, 1);      // relu
    gemm1(ffb, Wf2P[l], bf2 + l * HH, ff2b, FF_, HH, 0);

    hipLaunchKernelGGL(ln2_jk_kernel, dim3(NWAVE), dim3(256), 0, stream,
                       h, ff2b, ln2_g + l * HH, ln2_b + l * HH, h, xcb, l);
  }

  // ---- JK gate + softmax pooling ----
  gemm1(xcb, Wg1P, bg1, ghid, JK_, HH, 1);                  // relu(xc @ Wg1 + bg1)

  fill(gmb, -INFINITY, (size_t)GG);
  fill(gsb, 0.0f, (size_t)GG);
  fill(poolb, 0.0f, (size_t)GG * JK_);

  hipLaunchKernelGGL(gate_kernel, dim3(NWAVE), dim3(256), 0, stream,
                     ghid, Wg2, bg2, batch, gateb, gmb);
  hipLaunchKernelGGL(pool_kernel, dim3(NWAVE), dim3(256), 0, stream,
                     gateb, batch, gmb, gsb, xcb, poolb);

  // ---- final MLP head ----
  hipLaunchKernelGGL(head_kernel, dim3(GG), dim3(128), 0, stream,
                     poolb, gsb, u, Wh1, bh1, Wh2, bh2, out);
}
